// Attention_78202764525997
// MI455X (gfx1250) — matrix operand order repaired and checked by reading, not compiled, not run on hardware
//
#include <hip/hip_runtime.h>
#include <hip/hip_bf16.h>

// ---------------------------------------------------------------------------
// MI455X (gfx1250) attention block:
//   Q/K/V proj (bf16 WMMA GEMM) -> flash attention (bf16 WMMA, fp32 softmax,
//   TDM tensor_load_to_lds for K tiles) -> output proj + residual (bf16 WMMA
//   GEMM) -> in-place LayerNorm.
// ---------------------------------------------------------------------------

#define NQ     4096
#define NKEYS  4096
#define DMODEL 1024
#define NHEAD  16
#define DHEAD  64

typedef __attribute__((ext_vector_type(16))) __bf16 bf16x16;
typedef __attribute__((ext_vector_type(8)))  __bf16 bf16x8;
typedef __attribute__((ext_vector_type(8)))  float  f32x8;
typedef unsigned int v4u __attribute__((ext_vector_type(4)));
typedef int          v4i __attribute__((ext_vector_type(4)));
typedef int          v8i __attribute__((ext_vector_type(8)));

__device__ __forceinline__ f32x8 wmma_bf16(bf16x16 a, bf16x16 b, f32x8 c) {
  // D = A(16x32 bf16) * B(32x16 bf16) + C(16x16 f32)
  return __builtin_amdgcn_wmma_f32_16x16x32_bf16(
      /*neg_a=*/false, a, /*neg_b=*/false, b,
      /*c_mod=*/(short)0, c, /*reuse_a=*/false, /*reuse_b=*/false);
}

// ---------------------------------------------------------------------------
// Tensor Data Mover: DMA a 2D tile (tile_w x tile_h elements, 2 bytes each)
// from global (row stride row_stride_elems) into LDS at lds_addr.
// D# layout per CDNA5 ISA ch.8 (group0 128b, group1 256b).
// ---------------------------------------------------------------------------
#if __has_builtin(__builtin_amdgcn_tensor_load_to_lds)
#define HAVE_TDM 1
__device__ __forceinline__ void tdm_load_tile_2d(unsigned lds_addr,
                                                 const void* gptr,
                                                 unsigned tile_w,
                                                 unsigned tile_h,
                                                 unsigned row_stride_elems) {
  const unsigned long long ga = (unsigned long long)(size_t)gptr;
  v4u g0;
  g0[0] = 1u;                                    // count=1, user descriptor
  g0[1] = lds_addr;                              // LDS byte address
  g0[2] = (unsigned)(ga & 0xffffffffu);          // global_addr[31:0]
  g0[3] = (unsigned)((ga >> 32) & 0x01ffffffu)   // global_addr[56:32]
          | (2u << 30);                          // type=2 ("image")
  v8i g1;
  g1[0] = (int)(1u << 16);                       // data_size=1 (2B), wg_mask=0
  g1[1] = (int)((tile_w & 0xffffu) << 16);       // tensor_dim0[15:0] @63:48
  g1[2] = (int)(((tile_w >> 16) & 0xffffu)       // tensor_dim0[31:16] @79:64
          | ((tile_h & 0xffffu) << 16));         // tensor_dim1[15:0] @95:80
  g1[3] = (int)(((tile_h >> 16) & 0xffffu)       // tensor_dim1[31:16] @111:96
          | ((tile_w & 0xffffu) << 16));         // tile_dim0 @127:112
  g1[4] = (int)(tile_h & 0xffffu);               // tile_dim1 @143:128, tile_dim2=0
  g1[5] = (int)row_stride_elems;                 // tensor_dim0_stride[31:0]
  g1[6] = 0;                                     // stride hi, dim1_stride lo
  g1[7] = 0;
  const v4i z4 = {0, 0, 0, 0};
#if defined(__clang_major__) && (__clang_major__ >= 23)
  const v8i z8 = {0, 0, 0, 0, 0, 0, 0, 0};
  __builtin_amdgcn_tensor_load_to_lds(g0, g1, z4, z4, z8, 0);
#else
  __builtin_amdgcn_tensor_load_to_lds(g0, g1, z4, z4, 0);
#endif
}
#else
#define HAVE_TDM 0
#endif

// A-operand (16x32, 16-bit) layout: lane L holds row M=L%16; half h=L/16:
//   VGPR0-7 = K [h*16 .. h*16+15] (contiguous, matches load_b_frag)
__device__ __forceinline__ bf16x16 load_a_frag(const __bf16* base, int ldm,
                                               int r0, int k0, int lane) {
  const int r  = r0 + (lane & 15);
  const int hh = lane >> 4;
  const __bf16* p = base + r * ldm + k0 + hh * 16;
  bf16x8 lo = *(const bf16x8*)p;
  bf16x8 hi = *(const bf16x8*)(p + 8);
  bf16x16 a;
#pragma unroll
  for (int i = 0; i < 8; ++i) { a[i] = lo[i]; a[i + 8] = hi[i]; }
  return a;
}

// B-operand (32x16, 16-bit) layout: lane L holds col N=L%16; half h=L/16
// holds K [h*16 .. h*16+15] packed 2-per-VGPR. base is N-major (ld = ldm).
__device__ __forceinline__ bf16x16 load_b_frag(const __bf16* base, int ldm,
                                               int n0, int k0, int lane) {
  const int n  = n0 + (lane & 15);
  const int hh = lane >> 4;
  return *(const bf16x16*)(base + n * ldm + k0 + hh * 16);
}

// reductions inside a 16-lane half of a wave32 (xor masks stay in-half)
__device__ __forceinline__ float red_max16(float v) {
#pragma unroll
  for (int m = 8; m >= 1; m >>= 1) v = fmaxf(v, __shfl_xor(v, m, 32));
  return v;
}
__device__ __forceinline__ float red_sum16(float v) {
#pragma unroll
  for (int m = 8; m >= 1; m >>= 1) v += __shfl_xor(v, m, 32);
  return v;
}

// ---------------------------------------------------------------------------
// GEMM 1: C[n,j] = bf16( X[n,:] @ W[:,j] + b[j] )   (X,W fp32; out bf16)
// grid = (DMODEL/128, NQ/128, 3)  block = 256 (8 waves)
// ---------------------------------------------------------------------------
__global__ __launch_bounds__(256) void gemm_proj_kernel(
    const float* __restrict__ queries, const float* __restrict__ keys,
    const float* __restrict__ values,
    const float* __restrict__ Wq, const float* __restrict__ bq,
    const float* __restrict__ Wk, const float* __restrict__ bk,
    const float* __restrict__ Wv, const float* __restrict__ bv,
    __bf16* __restrict__ Qb, __bf16* __restrict__ Kb, __bf16* __restrict__ Vb) {
  const float* X; const float* W; const float* bias; __bf16* out;
  if (blockIdx.z == 0)      { X = queries; W = Wq; bias = bq; out = Qb; }
  else if (blockIdx.z == 1) { X = keys;    W = Wk; bias = bk; out = Kb; }
  else                      { X = values;  W = Wv; bias = bv; out = Vb; }

  __shared__ __attribute__((aligned(32))) __bf16 As[128 * 32];  // row-major
  __shared__ __attribute__((aligned(32))) __bf16 Bs[128 * 32];  // N-major

  const int t    = threadIdx.x;
  const int lane = t & 31;
  const int wave = t >> 5;
  const int h    = lane >> 4, lr = lane & 15;
  const int m0   = blockIdx.y * 128;
  const int n0   = blockIdx.x * 128;
  const int Moff = (wave & 3) * 32;   // 4 waves along M
  const int Noff = (wave >> 2) * 64;  // 2 waves along N

  f32x8 acc[2][4];
#pragma unroll
  for (int i = 0; i < 2; ++i)
#pragma unroll
    for (int j = 0; j < 4; ++j)
#pragma unroll
      for (int e = 0; e < 8; ++e) acc[i][j][e] = 0.0f;

  const int ar = t >> 1;          // A tile: row 0..127
  const int ac = (t & 1) * 16;    //          16 cols per thread
  const int wr = t >> 3;          // W tile: k-row 0..31
  const int wc = (t & 7) * 16;    //          16 cols per thread

  for (int kk = 0; kk < DMODEL; kk += 32) {
    __syncthreads();
    {  // X tile 128x32 fp32 -> bf16 row-major
      const float* src = X + (size_t)(m0 + ar) * DMODEL + kk + ac;
#pragma unroll
      for (int j = 0; j < 16; j += 4) {
        float4 v = *(const float4*)(src + j);
        As[ar * 32 + ac + j + 0] = (__bf16)v.x;
        As[ar * 32 + ac + j + 1] = (__bf16)v.y;
        As[ar * 32 + ac + j + 2] = (__bf16)v.z;
        As[ar * 32 + ac + j + 3] = (__bf16)v.w;
      }
    }
    {  // W tile 32x128 fp32 -> bf16, stored transposed: Bs[n][k]
      const float* src = W + (size_t)(kk + wr) * DMODEL + n0 + wc;
#pragma unroll
      for (int j = 0; j < 16; j += 4) {
        float4 v = *(const float4*)(src + j);
        Bs[(wc + j + 0) * 32 + wr] = (__bf16)v.x;
        Bs[(wc + j + 1) * 32 + wr] = (__bf16)v.y;
        Bs[(wc + j + 2) * 32 + wr] = (__bf16)v.z;
        Bs[(wc + j + 3) * 32 + wr] = (__bf16)v.w;
      }
    }
    __syncthreads();

    bf16x16 a[2], b[4];
#pragma unroll
    for (int tm = 0; tm < 2; ++tm) a[tm] = load_a_frag(As, 32, Moff + tm * 16, 0, lane);
#pragma unroll
    for (int tn = 0; tn < 4; ++tn) b[tn] = load_b_frag(Bs, 32, Noff + tn * 16, 0, lane);
#pragma unroll
    for (int tm = 0; tm < 2; ++tm)
#pragma unroll
      for (int tn = 0; tn < 4; ++tn)
        acc[tm][tn] = wmma_bf16(a[tm], b[tn], acc[tm][tn]);
  }

  // epilogue: C/D layout M = g + 8*h, N = lane%16
#pragma unroll
  for (int tm = 0; tm < 2; ++tm)
#pragma unroll
    for (int tn = 0; tn < 4; ++tn) {
      const int col = n0 + Noff + tn * 16 + lr;
      const float bb = bias[col];
#pragma unroll
      for (int g = 0; g < 8; ++g) {
        const int row = m0 + Moff + tm * 16 + g + 8 * h;
        out[(size_t)row * DMODEL + col] = (__bf16)(acc[tm][tn][g] + bb);
      }
    }
}

// ---------------------------------------------------------------------------
// Flash attention: one block = (head, 64 query rows), 4 waves x 16 rows.
// Streams 64-key tiles: K tile via TDM DMA, S = QK^T/8 -> online softmax
// -> O += P V.
// ---------------------------------------------------------------------------
__global__ __launch_bounds__(128) void flash_attn_kernel(
    const __bf16* __restrict__ Qb, const __bf16* __restrict__ Kb,
    const __bf16* __restrict__ Vb, __bf16* __restrict__ Ob) {
  __shared__ __attribute__((aligned(32))) __bf16 Qs[64 * 64];  // [qrow][d]
  __shared__ __attribute__((aligned(32))) __bf16 Ks[64 * 64];  // [key ][d]
  __shared__ __attribute__((aligned(32))) __bf16 Vs[64 * 64];  // [d][key]  (transposed)
  __shared__ __attribute__((aligned(32))) __bf16 Ps[64 * 64];  // [qrow][key], wave-private strips

  const int t    = threadIdx.x;
  const int lane = t & 31;
  const int wave = t >> 5;
  const int h    = lane >> 4, lr = lane & 15;
  const int q0   = blockIdx.x * 64;
  const int head = blockIdx.y;
  const int w16  = wave * 16;

  {  // load Q tile (64x64 bf16), 2 threads per row
    const int r = t >> 1, c0 = (t & 1) * 32;
    const __bf16* src = Qb + (size_t)(q0 + r) * DMODEL + head * DHEAD + c0;
    __bf16* dst = &Qs[r * 64 + c0];
#pragma unroll
    for (int j = 0; j < 4; ++j)
      ((uint4*)dst)[j] = ((const uint4*)src)[j];
  }
  __syncthreads();
  const bf16x16 qa0 = load_a_frag(Qs, 64, w16, 0, lane);
  const bf16x16 qa1 = load_a_frag(Qs, 64, w16, 32, lane);

  f32x8 o[4];
#pragma unroll
  for (int td = 0; td < 4; ++td)
#pragma unroll
    for (int e = 0; e < 8; ++e) o[td][e] = 0.0f;

  float m_[8], l_[8];
#pragma unroll
  for (int g = 0; g < 8; ++g) { m_[g] = -1e30f; l_[g] = 0.0f; }

  for (int kb = 0; kb < NKEYS; kb += 64) {
    __syncthreads();  // previous tile fully consumed by all waves

#if HAVE_TDM
    if (wave == 0) {  // one TDM DMA per block: 64x64 bf16 K tile -> LDS
      tdm_load_tile_2d((unsigned)(size_t)(void*)Ks,
                       Kb + (size_t)kb * DMODEL + head * DHEAD,
                       /*tile_w=*/DHEAD, /*tile_h=*/64,
                       /*row_stride=*/DMODEL);
    }
#else
    {  // K tile row-major (fallback path)
      const int r = t >> 1, c0 = (t & 1) * 32;
      const __bf16* src = Kb + (size_t)(kb + r) * DMODEL + head * DHEAD + c0;
      __bf16* dst = &Ks[r * 64 + c0];
#pragma unroll
      for (int j = 0; j < 4; ++j)
        ((uint4*)dst)[j] = ((const uint4*)src)[j];
    }
#endif
    {  // V tile transposed into Vs[d][key] (vector path, overlaps the DMA)
      const int r = t >> 1, c0 = (t & 1) * 32;
      const __bf16* src = Vb + (size_t)(kb + r) * DMODEL + head * DHEAD + c0;
#pragma unroll
      for (int j = 0; j < 32; ++j) Vs[(c0 + j) * 64 + r] = src[j];
    }
    if (kb + 64 < NKEYS) {  // hint next tiles toward L2 (global_prefetch_b8)
      __builtin_prefetch(Kb + (size_t)(kb + 64) * DMODEL + head * DHEAD, 0, 1);
      __builtin_prefetch(Vb + (size_t)(kb + 64) * DMODEL + head * DHEAD, 0, 1);
    }
#if HAVE_TDM
    if (wave == 0) __builtin_amdgcn_s_wait_tensorcnt(0);  // DMA complete
#endif
    __syncthreads();  // all waves see finished K (TDM) + V tiles

    // S strip (16 rows x 64 keys) = Q K^T : B operand = K rows (N-major!)
    f32x8 s[4];
#pragma unroll
    for (int tn = 0; tn < 4; ++tn) {
#pragma unroll
      for (int e = 0; e < 8; ++e) s[tn][e] = 0.0f;
      bf16x16 b0 = load_b_frag(Ks, 64, tn * 16, 0, lane);
      bf16x16 b1 = load_b_frag(Ks, 64, tn * 16, 32, lane);
      s[tn] = wmma_bf16(qa0, b0, s[tn]);
      s[tn] = wmma_bf16(qa1, b1, s[tn]);
    }

    // scale + online softmax (per row g: M = g + 8*h)
    float rmax[8], esc[8], rsum[8];
#pragma unroll
    for (int g = 0; g < 8; ++g) rmax[g] = -1e30f;
#pragma unroll
    for (int tn = 0; tn < 4; ++tn)
#pragma unroll
      for (int g = 0; g < 8; ++g) {
        s[tn][g] *= 0.125f;  // 1/sqrt(64)
        rmax[g] = fmaxf(rmax[g], s[tn][g]);
      }
#pragma unroll
    for (int g = 0; g < 8; ++g) {
      rmax[g] = red_max16(rmax[g]);
      const float mn = fmaxf(m_[g], rmax[g]);
      esc[g] = __expf(m_[g] - mn);
      m_[g] = mn;
      rsum[g] = 0.0f;
    }
#pragma unroll
    for (int tn = 0; tn < 4; ++tn)
#pragma unroll
      for (int g = 0; g < 8; ++g) {
        const float p = __expf(s[tn][g] - m_[g]);
        rsum[g] += p;
        Ps[(w16 + g + 8 * h) * 64 + tn * 16 + lr] = (__bf16)p;
      }
#pragma unroll
    for (int g = 0; g < 8; ++g) {
      rsum[g] = red_sum16(rsum[g]);
      l_[g] = l_[g] * esc[g] + rsum[g];
    }
#pragma unroll
    for (int td = 0; td < 4; ++td)
#pragma unroll
      for (int g = 0; g < 8; ++g) o[td][g] *= esc[g];

    // O += P * V  (A = P strip from LDS; B = Vs, N-major in d)
    const bf16x16 pa0 = load_a_frag(Ps, 64, w16, 0, lane);
    const bf16x16 pa1 = load_a_frag(Ps, 64, w16, 32, lane);
#pragma unroll
    for (int td = 0; td < 4; ++td) {
      bf16x16 vb0 = load_b_frag(Vs, 64, td * 16, 0, lane);
      bf16x16 vb1 = load_b_frag(Vs, 64, td * 16, 32, lane);
      o[td] = wmma_bf16(pa0, vb0, o[td]);
      o[td] = wmma_bf16(pa1, vb1, o[td]);
    }
  }

  // normalize and store bf16 attention output
#pragma unroll
  for (int td = 0; td < 4; ++td)
#pragma unroll
    for (int g = 0; g < 8; ++g) {
      const int row = q0 + w16 + g + 8 * h;
      const int d   = td * 16 + lr;
      const float inv = 1.0f / l_[g];
      Ob[(size_t)row * DMODEL + head * DHEAD + d] = (__bf16)(o[td][g] * inv);
    }
}

// ---------------------------------------------------------------------------
// GEMM 2: out[n,j] = queries[n,j] + attn[n,:] @ Wo[:,j] + bo[j]   (fp32 out)
// ---------------------------------------------------------------------------
__global__ __launch_bounds__(256) void gemm_out_kernel(
    const __bf16* __restrict__ Aattn, const float* __restrict__ Wo,
    const float* __restrict__ bo, const float* __restrict__ queries,
    float* __restrict__ out) {
  __shared__ __attribute__((aligned(32))) __bf16 As[128 * 32];
  __shared__ __attribute__((aligned(32))) __bf16 Bs[128 * 32];

  const int t    = threadIdx.x;
  const int lane = t & 31;
  const int wave = t >> 5;
  const int h    = lane >> 4, lr = lane & 15;
  const int m0   = blockIdx.y * 128;
  const int n0   = blockIdx.x * 128;
  const int Moff = (wave & 3) * 32;
  const int Noff = (wave >> 2) * 64;

  f32x8 acc[2][4];
#pragma unroll
  for (int i = 0; i < 2; ++i)
#pragma unroll
    for (int j = 0; j < 4; ++j)
#pragma unroll
      for (int e = 0; e < 8; ++e) acc[i][j][e] = 0.0f;

  const int ar = t >> 1;
  const int ac = (t & 1) * 16;
  const int wr = t >> 3;
  const int wc = (t & 7) * 16;

  for (int kk = 0; kk < DMODEL; kk += 32) {
    __syncthreads();
    {  // bf16 A tile: two 16B vector copies per thread
      const __bf16* src = Aattn + (size_t)(m0 + ar) * DMODEL + kk + ac;
      __bf16* dst = &As[ar * 32 + ac];
      ((uint4*)dst)[0] = ((const uint4*)src)[0];
      ((uint4*)dst)[1] = ((const uint4*)src)[1];
    }
    {  // fp32 Wo tile -> bf16, transposed N-major
      const float* src = Wo + (size_t)(kk + wr) * DMODEL + n0 + wc;
#pragma unroll
      for (int j = 0; j < 16; j += 4) {
        float4 v = *(const float4*)(src + j);
        Bs[(wc + j + 0) * 32 + wr] = (__bf16)v.x;
        Bs[(wc + j + 1) * 32 + wr] = (__bf16)v.y;
        Bs[(wc + j + 2) * 32 + wr] = (__bf16)v.z;
        Bs[(wc + j + 3) * 32 + wr] = (__bf16)v.w;
      }
    }
    __syncthreads();

    bf16x16 a[2], b[4];
#pragma unroll
    for (int tm = 0; tm < 2; ++tm) a[tm] = load_a_frag(As, 32, Moff + tm * 16, 0, lane);
#pragma unroll
    for (int tn = 0; tn < 4; ++tn) b[tn] = load_b_frag(Bs, 32, Noff + tn * 16, 0, lane);
#pragma unroll
    for (int tm = 0; tm < 2; ++tm)
#pragma unroll
      for (int tn = 0; tn < 4; ++tn)
        acc[tm][tn] = wmma_bf16(a[tm], b[tn], acc[tm][tn]);
  }

#pragma unroll
  for (int tm = 0; tm < 2; ++tm)
#pragma unroll
    for (int tn = 0; tn < 4; ++tn) {
      const int col = n0 + Noff + tn * 16 + lr;
      const float bb = bo[col];
#pragma unroll
      for (int g = 0; g < 8; ++g) {
        const int row = m0 + Moff + tm * 16 + g + 8 * h;
        out[(size_t)row * DMODEL + col] =
            acc[tm][tn][g] + bb + queries[(size_t)row * DMODEL + col];
      }
    }
}

// ---------------------------------------------------------------------------
// In-place LayerNorm over DMODEL=1024, eps=1e-12. One block per row.
// ---------------------------------------------------------------------------
__global__ __launch_bounds__(256) void layernorm_kernel(
    float* __restrict__ out, const float* __restrict__ gamma,
    const float* __restrict__ beta) {
  __shared__ float red[256];
  const int row = blockIdx.x;
  const int t   = threadIdx.x;
  float* x = out + (size_t)row * DMODEL;

  const int c = t * 4;
  float4 v = *(const float4*)(x + c);
  red[t] = v.x + v.y + v.z + v.w;
  __syncthreads();
#pragma unroll
  for (int off = 128; off > 0; off >>= 1) {
    if (t < off) red[t] += red[t + off];
    __syncthreads();
  }
  const float mu = red[0] * (1.0f / DMODEL);
  __syncthreads();

  const float dx = v.x - mu, dy = v.y - mu, dz = v.z - mu, dw = v.w - mu;
  red[t] = dx * dx + dy * dy + dz * dz + dw * dw;
  __syncthreads();
#pragma unroll
  for (int off = 128; off > 0; off >>= 1) {
    if (t < off) red[t] += red[t + off];
    __syncthreads();
  }
  const float rs = rsqrtf(red[0] * (1.0f / DMODEL) + 1e-12f);

  float4 g4 = *(const float4*)(gamma + c);
  float4 b4 = *(const float4*)(beta + c);
  float4 o4;
  o4.x = dx * rs * g4.x + b4.x;
  o4.y = dy * rs * g4.y + b4.y;
  o4.z = dz * rs * g4.z + b4.z;
  o4.w = dw * rs * g4.w + b4.w;
  *(float4*)(x + c) = o4;
}

// ---------------------------------------------------------------------------
extern "C" void kernel_launch(void* const* d_in, const int* in_sizes, int n_in,
                              void* d_out, int out_size, void* d_ws, size_t ws_size,
                              hipStream_t stream) {
  (void)in_sizes; (void)n_in; (void)out_size; (void)ws_size;

  const float* queries = (const float*)d_in[0];
  const float* keys    = (const float*)d_in[1];
  const float* values  = (const float*)d_in[2];
  const float* Wq      = (const float*)d_in[3];
  const float* bq      = (const float*)d_in[4];
  const float* Wk      = (const float*)d_in[5];
  const float* bk      = (const float*)d_in[6];
  const float* Wv      = (const float*)d_in[7];
  const float* bv      = (const float*)d_in[8];
  const float* Wo      = (const float*)d_in[9];
  const float* bo      = (const float*)d_in[10];
  const float* gamma   = (const float*)d_in[11];
  const float* beta    = (const float*)d_in[12];
  float* out = (float*)d_out;

  // workspace: 4 x (4096*1024) bf16 = 32 MB
  __bf16* Qb = (__bf16*)d_ws;
  __bf16* Kb = Qb + (size_t)NQ * DMODEL;
  __bf16* Vb = Kb + (size_t)NKEYS * DMODEL;
  __bf16* Ob = Vb + (size_t)NKEYS * DMODEL;

  gemm_proj_kernel<<<dim3(DMODEL / 128, NQ / 128, 3), 256, 0, stream>>>(
      queries, keys, values, Wq, bq, Wk, bk, Wv, bv, Qb, Kb, Vb);

  flash_attn_kernel<<<dim3(NQ / 64, NHEAD), 128, 0, stream>>>(Qb, Kb, Vb, Ob);

  gemm_out_kernel<<<dim3(DMODEL / 128, NQ / 128), 256, 0, stream>>>(
      Ob, Wo, bo, queries, out);

  layernorm_kernel<<<NQ, 256, 0, stream>>>(out, gamma, beta);
}
